// SelfAttention_24343874633921
// MI455X (gfx1250) — matrix operand order repaired and checked by reading, not compiled, not run on hardware
//
#include <hip/hip_runtime.h>
#include <hip/hip_bf16.h>
#include <cstdint>

#define DIM   768
#define NHEAD 12
#define HDIM  64
#define SEQ   2048
#define BATCH 4
#define ROWS  (BATCH*SEQ)      // 8192
#define C3    (3*DIM)          // 2304
#define ATT_SCALE 0.125f       // 1/sqrt(64)
#define LN_EPS 1e-5f

typedef __attribute__((ext_vector_type(16))) __bf16 bf16x16;
typedef __attribute__((ext_vector_type(8)))  __bf16 bf16x8;
typedef __attribute__((ext_vector_type(8)))  float  f32x8;

__device__ __forceinline__ int lane_id() { return (int)(threadIdx.x & 31u); }

// ---- CDNA5 async global->LDS copy (ASYNCcnt-tracked, no VGPR data path) ----
// LDS offset = low 32 bits of generic LDS ptr.
__device__ __forceinline__ uint32_t lds_off(const void* p) {
  return (uint32_t)(uintptr_t)p;
}
__device__ __forceinline__ void async_copy_b128(const void* lds_dst, const void* gsrc) {
  asm volatile("global_load_async_to_lds_b128 %0, %1, off"
               :: "v"(lds_off(lds_dst)), "v"((uint64_t)(uintptr_t)gsrc)
               : "memory");
}
__device__ __forceinline__ void wait_async0() {
  asm volatile("s_wait_asynccnt 0x0" ::: "memory");
}

// 16-bit A-matrix 16x32 fragment: lane m=lane&15,
// elems 0..15 = K[(lane>>4)*16 .. +15] (contiguous, matches B fragments)
__device__ __forceinline__ bf16x16 frag_A16x32(const __bf16* base, int ld) {
  int lane = lane_id();
  int m  = lane & 15;
  int kb = (lane >> 4) * 16;
  const __bf16* p = base + m * ld + kb;
  bf16x8 lo = *(const bf16x8*)p;
  bf16x8 hi = *(const bf16x8*)(p + 8);
  return __builtin_shufflevector(lo, hi, 0,1,2,3,4,5,6,7,8,9,10,11,12,13,14,15);
}

// 32 contiguous bytes -> one fragment register set (two ds_load_b128)
__device__ __forceinline__ bf16x16 frag_contig32(const __bf16* p) {
  bf16x8 lo = *(const bf16x8*)p;
  bf16x8 hi = *(const bf16x8*)(p + 8);
  return __builtin_shufflevector(lo, hi, 0,1,2,3,4,5,6,7,8,9,10,11,12,13,14,15);
}

// ---------------- weight conversion f32 -> bf16 ----------------
__global__ void convert_w_kernel(const float* __restrict__ qkv_w,
                                 const float* __restrict__ proj_w,
                                 __bf16* __restrict__ qkv_bf,
                                 __bf16* __restrict__ proj_bf) {
  int i = blockIdx.x * blockDim.x + threadIdx.x;
  int stride = gridDim.x * blockDim.x;
  for (int j = i; j < DIM * C3; j += stride)  qkv_bf[j]  = (__bf16)qkv_w[j];
  for (int j = i; j < DIM * DIM; j += stride) proj_bf[j] = (__bf16)proj_w[j];
}

// ---------------- LayerNorm + pos_embed -> bf16 ----------------
__global__ __launch_bounds__(256) void ln_pos_kernel(
    const float* __restrict__ x, const float* __restrict__ pos,
    const float* __restrict__ g, const float* __restrict__ b,
    __bf16* __restrict__ xp) {
  int row = blockIdx.x;
  const float* xr = x   + (size_t)row * DIM;
  const float* pr = pos + (size_t)row * DIM;
  int t = threadIdx.x;
  float v0 = xr[t], v1 = xr[t + 256], v2 = xr[t + 512];
  float s  = v0 + v1 + v2;
  float ss = v0 * v0 + v1 * v1 + v2 * v2;
#pragma unroll
  for (int o = 16; o; o >>= 1) {
    s  += __shfl_down(s,  o, 32);
    ss += __shfl_down(ss, o, 32);
  }
  __shared__ float sh_s[8], sh_ss[8], sh_stats[2];
  int w = t >> 5;
  if (lane_id() == 0) { sh_s[w] = s; sh_ss[w] = ss; }
  __syncthreads();
  if (t == 0) {
    float S = 0.f, SS = 0.f;
    for (int i = 0; i < 8; i++) { S += sh_s[i]; SS += sh_ss[i]; }
    float mu  = S / (float)DIM;
    float var = SS / (float)DIM - mu * mu;
    sh_stats[0] = mu;
    sh_stats[1] = rsqrtf(var + LN_EPS);
  }
  __syncthreads();
  float mu = sh_stats[0], rs = sh_stats[1];
  __bf16* xo = xp + (size_t)row * DIM;
  xo[t]       = (__bf16)((v0 - mu) * rs * g[t]       + b[t]       + pr[t]);
  xo[t + 256] = (__bf16)((v1 - mu) * rs * g[t + 256] + b[t + 256] + pr[t + 256]);
  xo[t + 512] = (__bf16)((v2 - mu) * rs * g[t + 512] + b[t + 512] + pr[t + 512]);
}

// ---------------- tiled bf16 WMMA GEMM ----------------
// Block tile 64(M) x 128(N), BK=32, 256 threads = 8 waves.
// A tile staged via async global->LDS; B tile staged TRANSPOSED (ld=40, conflict-free)
// so B-fragments are contiguous ds_load_b128 pairs.
// MODE 0: out = A*B + bias (f32).  MODE 1: scatter QKV into q(*scale)/k/v [B,H,K,D] bf16.
#define BLD 40
template<int MODE>
__global__ __launch_bounds__(256) void gemm_bf16_kernel(
    const __bf16* __restrict__ A, const __bf16* __restrict__ Bm,
    const float* __restrict__ bias, int N, int Kc,
    float* __restrict__ outF,
    __bf16* __restrict__ qd, __bf16* __restrict__ kd, __bf16* __restrict__ vd) {
  __shared__ __align__(32) __bf16 As[64 * 32];
  __shared__ __align__(32) __bf16 BsT[128 * BLD];   // [n][k] transposed
  int tid = threadIdx.x;
  int m0 = blockIdx.y * 64, n0 = blockIdx.x * 128;
  int w = tid >> 5, wm = w & 3, wn = w >> 2;
  int lane = lane_id();
  int n_ = lane & 15, rh = (lane >> 4) * 8, kb16 = (lane >> 4) * 16;
  f32x8 acc[4] = {};

  for (int k0 = 0; k0 < Kc; k0 += 32) {
    __syncthreads();
    {  // A tile 64x32: one async b128 per thread
      int r = tid >> 2, c = (tid & 3) * 8;
      async_copy_b128(&As[r * 32 + c], &A[(size_t)(m0 + r) * Kc + k0 + c]);
    }
#pragma unroll
    for (int j = 0; j < 2; j++) {  // B tile 32x128 -> BsT transpose
      int idx = tid * 2 + j;
      int r = idx >> 4, c = (idx & 15) * 8;
      bf16x8 bv = *(const bf16x8*)&Bm[(size_t)(k0 + r) * N + n0 + c];
#pragma unroll
      for (int j2 = 0; j2 < 8; j2++) BsT[(c + j2) * BLD + r] = bv[j2];
    }
    wait_async0();
    __syncthreads();
    bf16x16 a = frag_A16x32(&As[wm * 16 * 32], 32);
#pragma unroll
    for (int t = 0; t < 4; t++) {
      bf16x16 bb = frag_contig32(&BsT[(wn * 64 + t * 16 + n_) * BLD + kb16]);
      acc[t] = __builtin_amdgcn_wmma_f32_16x16x32_bf16(
          false, a, false, bb, (short)0, acc[t], false, false);
    }
  }

#pragma unroll
  for (int t = 0; t < 4; t++) {
#pragma unroll
    for (int r = 0; r < 8; r++) {
      int grow = m0 + wm * 16 + rh + r;
      int gcol = n0 + wn * 64 + t * 16 + n_;
      float valf = acc[t][r] + bias[gcol];
      if (MODE == 0) {
        outF[(size_t)grow * N + gcol] = valf;
      } else {
        int t3 = gcol / DIM, rem = gcol % DIM;
        int h = rem >> 6, d = rem & 63;
        int bi = grow >> 11, seq = grow & 2047;
        size_t di = (((size_t)bi * NHEAD + h) * SEQ + seq) * HDIM + d;
        if (t3 == 0)      qd[di] = (__bf16)(valf * ATT_SCALE);
        else if (t3 == 1) kd[di] = (__bf16)valf;
        else              vd[di] = (__bf16)valf;
      }
    }
  }
}

// ---------------- flash attention ----------------
// Grid: (SEQ/64, NHEAD, BATCH); 128 threads = 4 waves; wave w owns 16 q-rows.
// q,k staged via async global->LDS (row-major is fragment-friendly for both);
// v staged transposed (ld=136, conflict-free) for contiguous B-fragments.
#define VLD 136
__global__ __launch_bounds__(128) void attention_kernel(
    const __bf16* __restrict__ q, const __bf16* __restrict__ k,
    const __bf16* __restrict__ v, const float* __restrict__ rel,
    __bf16* __restrict__ attn_out) {
  __shared__ __align__(32) __bf16 qs[64 * 64];
  __shared__ __align__(32) __bf16 ks[128 * 64];
  __shared__ __align__(32) __bf16 vsT[64 * VLD];     // [d][kv]
  __shared__ __align__(32) __bf16 ps[4 * 16 * 128];  // per-wave P tiles
  int tid = threadIdx.x;
  int w = tid >> 5, lane = tid & 31;
  int n_ = lane & 15, rh = (lane >> 4) * 8, kb16 = (lane >> 4) * 16;
  int b = blockIdx.z, h = blockIdx.y, q0 = blockIdx.x * 64;
  size_t headoff = ((size_t)b * NHEAD + h) * (size_t)SEQ * HDIM;
  const __bf16* qg = q + headoff;
  const __bf16* kg = k + headoff;
  const __bf16* vg = v + headoff;

  for (int i = tid; i < 512; i += 128) {  // stage q tile 64x64 (async)
    int r = i >> 3, c = (i & 7) * 8;
    async_copy_b128(&qs[r * 64 + c], &qg[(size_t)(q0 + r) * 64 + c]);
  }

  f32x8 o[4] = {};
  float mrow[8], lrow[8];
#pragma unroll
  for (int r = 0; r < 8; r++) { mrow[r] = -INFINITY; lrow[r] = 0.f; }
  const float* relb = rel + ((size_t)b * SEQ + q0 + w * 16 + rh) * SEQ;
  __bf16* pw = &ps[w * 16 * 128];

  for (int kv0 = 0; kv0 < SEQ; kv0 += 128) {
    __syncthreads();
    for (int i = tid; i < 1024; i += 128) {  // stage k (async) + v (transposed)
      int r = i >> 3, c = (i & 7) * 8;
      async_copy_b128(&ks[r * 64 + c], &kg[(size_t)(kv0 + r) * 64 + c]);
      bf16x8 vv = *(const bf16x8*)&vg[(size_t)(kv0 + r) * 64 + c];
#pragma unroll
      for (int j = 0; j < 8; j++) vsT[(c + j) * VLD + r] = vv[j];
    }
    if (kv0 + 128 < SEQ) {  // prefetch next tile's rel_pos_bias rows
#pragma unroll
      for (int r = 0; r < 8; r++)
        __builtin_prefetch(&relb[(size_t)r * SEQ + kv0 + 128], 0, 1);
    }
    wait_async0();
    __syncthreads();

    // S = q * k^T  (q pre-scaled).  B-frag of k^T is contiguous along D.
    f32x8 s[8] = {};
#pragma unroll
    for (int dstep = 0; dstep < 64; dstep += 32) {
      bf16x16 a = frag_A16x32(&qs[w * 16 * 64 + dstep], 64);
#pragma unroll
      for (int t = 0; t < 8; t++) {
        bf16x16 bb = frag_contig32(&ks[(t * 16 + n_) * 64 + dstep + kb16]);
        s[t] = __builtin_amdgcn_wmma_f32_16x16x32_bf16(
            false, a, false, bb, (short)0, s[t], false, false);
      }
    }
    // + rel_pos_bias (streamed once from HBM, lanes coalesce along n_)
#pragma unroll
    for (int t = 0; t < 8; t++)
#pragma unroll
      for (int r = 0; r < 8; r++)
        s[t][r] += relb[(size_t)r * SEQ + kv0 + t * 16 + n_];

    // online softmax: rows live across the 16-lane half-wave group
    float alpha[8];
#pragma unroll
    for (int r = 0; r < 8; r++) {
      float mx = mrow[r];
#pragma unroll
      for (int t = 0; t < 8; t++) mx = fmaxf(mx, s[t][r]);
#pragma unroll
      for (int msk = 1; msk < 16; msk <<= 1)
        mx = fmaxf(mx, __shfl_xor(mx, msk, 32));
      alpha[r] = __expf(mrow[r] - mx);
      mrow[r] = mx;
    }
#pragma unroll
    for (int ot = 0; ot < 4; ot++)
#pragma unroll
      for (int r = 0; r < 8; r++) o[ot][r] *= alpha[r];

    float lsum[8];
#pragma unroll
    for (int r = 0; r < 8; r++) lsum[r] = 0.f;
#pragma unroll
    for (int t = 0; t < 8; t++)
#pragma unroll
      for (int r = 0; r < 8; r++) {
        float p = __expf(s[t][r] - mrow[r]);
        lsum[r] += p;
        pw[(rh + r) * 128 + t * 16 + n_] = (__bf16)p;  // re-layout C/D -> A via LDS
      }
#pragma unroll
    for (int r = 0; r < 8; r++) {
      float sv = lsum[r];
#pragma unroll
      for (int msk = 1; msk < 16; msk <<= 1) sv += __shfl_xor(sv, msk, 32);
      lrow[r] = lrow[r] * alpha[r] + sv;
    }

    // O += P * V   (V fragments contiguous from vsT)
#pragma unroll
    for (int ks2 = 0; ks2 < 4; ks2++) {
      bf16x16 a = frag_A16x32(&pw[ks2 * 32], 128);
#pragma unroll
      for (int ot = 0; ot < 4; ot++) {
        bf16x16 bb = frag_contig32(&vsT[(ot * 16 + n_) * VLD + ks2 * 32 + kb16]);
        o[ot] = __builtin_amdgcn_wmma_f32_16x16x32_bf16(
            false, a, false, bb, (short)0, o[ot], false, false);
      }
    }
  }

  // normalize and write [B,K,C] bf16
  __bf16* outp = attn_out + ((size_t)b * SEQ + q0 + w * 16 + rh) * DIM + h * HDIM;
#pragma unroll
  for (int ot = 0; ot < 4; ot++)
#pragma unroll
    for (int r = 0; r < 8; r++)
      outp[(size_t)r * DIM + ot * 16 + n_] = (__bf16)(o[ot][r] / lrow[r]);
}

// ---------------- host launcher ----------------
extern "C" void kernel_launch(void* const* d_in, const int* in_sizes, int n_in,
                              void* d_out, int out_size, void* d_ws, size_t ws_size,
                              hipStream_t stream) {
  const float* x      = (const float*)d_in[0];
  const float* pos    = (const float*)d_in[1];
  const float* rel    = (const float*)d_in[2];
  const float* ln_g   = (const float*)d_in[3];
  const float* ln_b   = (const float*)d_in[4];
  const float* qkv_w  = (const float*)d_in[5];
  const float* qkv_b  = (const float*)d_in[6];
  const float* proj_w = (const float*)d_in[7];
  const float* proj_b = (const float*)d_in[8];

  char* ws = (char*)d_ws;
  __bf16* qkv_wbf  = (__bf16*)(ws + 0);          // 768*2304*2   = 3,538,944 B
  __bf16* proj_wbf = (__bf16*)(ws + 3538944);    // 768*768*2    = 1,179,648 B
  __bf16* xp       = (__bf16*)(ws + 4718592);    // 8192*768*2   = 12,582,912 B
  __bf16* qb       = (__bf16*)(ws + 17301504);   // [B,H,K,D] bf16
  __bf16* kb       = (__bf16*)(ws + 29884416);
  __bf16* vb       = (__bf16*)(ws + 42467328);
  __bf16* ao       = (__bf16*)(ws + 55050240);   // attn out [B,K,C] bf16
  // total 67,633,152 B

  convert_w_kernel<<<512, 256, 0, stream>>>(qkv_w, proj_w, qkv_wbf, proj_wbf);
  ln_pos_kernel<<<ROWS, 256, 0, stream>>>(x, pos, ln_g, ln_b, xp);
  gemm_bf16_kernel<1><<<dim3(C3 / 128, ROWS / 64), 256, 0, stream>>>(
      xp, qkv_wbf, qkv_b, C3, DIM, nullptr, qb, kb, vb);
  attention_kernel<<<dim3(SEQ / 64, NHEAD, BATCH), 128, 0, stream>>>(
      qb, kb, vb, rel, ao);
  gemm_bf16_kernel<0><<<dim3(DIM / 128, ROWS / 64), 256, 0, stream>>>(
      ao, proj_wbf, proj_b, DIM, DIM, (float*)d_out, nullptr, nullptr, nullptr);
}
